// IntraTemporalAttention_68350109549173
// MI455X (gfx1250) — matrix operand order repaired and checked by reading, not compiled, not run on hardware
//
#include <hip/hip_runtime.h>
#include <hip/hip_bf16.h>

// ---------------------------------------------------------------------------
// Problem constants (match reference)
// ---------------------------------------------------------------------------
constexpr int kB   = 32;     // batch
constexpr int kS   = 8192;   // max_enc_steps
constexpr int kH2  = 256;    // 2*hidden
constexpr int kNBLK = 32;    // s-chunks per batch
constexpr int kCHUNK = kS / kNBLK;   // 256 s per block

typedef __attribute__((ext_vector_type(16))) _Float16 v16h;
typedef __attribute__((ext_vector_type(8)))  float    v8f;

// ---------------------------------------------------------------------------
// Kernel 1: q = h_d_t @ W  via V_WMMA_F32_16X16X32_F16
// Grid: 32 blocks (one wave each). Tile (m0,n0) of the 32x256 output.
// K = 256 -> 8 WMMA steps of K=32.
// ---------------------------------------------------------------------------
__global__ __launch_bounds__(32)
void qgemm_wmma(const float* __restrict__ hdt,   // (32,256)
                const float* __restrict__ W,     // (256,256)
                float* __restrict__ q)           // (32,256)
{
    const int tile = blockIdx.x;         // 0..31
    const int m0   = (tile >> 4) << 4;   // 0 or 16
    const int n0   = (tile & 15) << 4;   // 0..240
    const int lane = threadIdx.x;        // 0..31 (wave32)
    const int mr   = lane & 15;
    const int hi   = lane >> 4;          // half-wave select

    v8f c = {};
    const int khalfA = hi * 8;           // A: lanes16-31 hold K+8 group
    const int kbB    = hi * 8;           // B: lanes16-31 hold K+8 group

    for (int k0 = 0; k0 < kH2; k0 += 32) {
        v16h a, b;
        const float* arow = hdt + (m0 + mr) * kH2 + k0;
        #pragma unroll
        for (int j = 0; j < 16; ++j) {
            const int v = j >> 1, odd = j & 1;
            const int K = (v < 4 ? (khalfA + 2 * v)
                                 : (16 + khalfA + 2 * (v - 4))) + odd;
            a[j] = (_Float16)arow[K];
        }
        const float* bcol = W + (size_t)(k0 + kbB) * kH2 + (n0 + mr);
        #pragma unroll
        for (int j = 0; j < 16; ++j) {
            b[j] = (_Float16)bcol[(size_t)(j < 8 ? j : j + 8) * kH2];
        }
        // (neg_a, A, neg_b, B, c_mod, C, reuse_a, reuse_b)
        c = __builtin_amdgcn_wmma_f32_16x16x32_f16(false, a, false, b,
                                                   (short)0, c, false, false);
    }

    #pragma unroll
    for (int v = 0; v < 8; ++v) {
        const int M = m0 + v + hi * 8;
        q[M * kH2 + (n0 + mr)] = c[v];
    }
}

// ---------------------------------------------------------------------------
// Kernel 2: single streaming pass over h_enc (the 256 MiB tensor).
// Grid: (kNBLK, kB) blocks x 256 threads (8 waves).
// Each wave: one 1KB row at a time, lane holds 8 contiguous dims (2x b128).
// Produces: unnormalized alphat (w), new_sum (final), per-block ct/norm
// partials in workspace.
// ---------------------------------------------------------------------------
__global__ __launch_bounds__(256)
void attn_stream(const float* __restrict__ q,          // (32,256)
                 const float* __restrict__ h_enc,      // (32,8192,256)
                 const float* __restrict__ mask,       // (32,8192)
                 const float* __restrict__ sumexp,     // (32,8192)
                 float* __restrict__ alphat_out,       // (32,8192) unnormalized
                 float* __restrict__ nsum_out,         // (32,8192)
                 float* __restrict__ partial_ct,       // (32,kNBLK,256)
                 float* __restrict__ partial_norm)     // (32,kNBLK)
{
    const int blk  = blockIdx.x;     // s-chunk
    const int b    = blockIdx.y;     // batch
    const int tid  = threadIdx.x;
    const int lane = tid & 31;
    const int wave = tid >> 5;       // 0..7

    __shared__ float lds_ct[8][kH2];
    __shared__ float lds_norm[8];

    // Per-lane slice of q[b]: dims [lane*8, lane*8+8)
    const float* qb = q + b * kH2 + lane * 8;
    const float4 q0 = *(const float4*)(qb);
    const float4 q1 = *(const float4*)(qb + 4);

    float4 acc0 = make_float4(0.f, 0.f, 0.f, 0.f);
    float4 acc1 = make_float4(0.f, 0.f, 0.f, 0.f);
    float  norm_acc = 0.f;

    const int    s_base = blk * kCHUNK;
    const float* enc_b  = h_enc + (size_t)b * kS * kH2;

    for (int i = 0; i < kCHUNK / 8; ++i) {
        const int    s   = s_base + i * 8 + wave;
        const float* row = enc_b + (size_t)s * kH2 + lane * 8;

        // prefetch this wave's next row (s+8) -> global_prefetch_b8
        __builtin_prefetch(row + 8 * kH2, 0, 0);

        const float4 r0 = *(const float4*)(row);
        const float4 r1 = *(const float4*)(row + 4);

        // partial dot over this lane's 8 dims
        float p = r0.x * q0.x + r0.y * q0.y + r0.z * q0.z + r0.w * q0.w
                + r1.x * q1.x + r1.y * q1.y + r1.z * q1.z + r1.w * q1.w;

        // wave32 butterfly reduction -> all lanes hold the full score
        #pragma unroll
        for (int off = 16; off > 0; off >>= 1)
            p += __shfl_xor(p, off, 32);

        const float m  = mask[(size_t)b * kS + s];
        const float se = sumexp[(size_t)b * kS + s];
        const float e  = __expf(p);
        const float w  = e * m / se;           // unnormalized attention weight

        if (lane == 0) {
            nsum_out[(size_t)b * kS + s]   = se + e;
            alphat_out[(size_t)b * kS + s] = w;
        }

        // context accumulation reuses the registers already loaded
        acc0.x += w * r0.x; acc0.y += w * r0.y;
        acc0.z += w * r0.z; acc0.w += w * r0.w;
        acc1.x += w * r1.x; acc1.y += w * r1.y;
        acc1.z += w * r1.z; acc1.w += w * r1.w;
        norm_acc += w;
    }

    // Stage wave partials in LDS and reduce across the 8 waves
    float* dst = &lds_ct[wave][lane * 8];
    dst[0] = acc0.x; dst[1] = acc0.y; dst[2] = acc0.z; dst[3] = acc0.w;
    dst[4] = acc1.x; dst[5] = acc1.y; dst[6] = acc1.z; dst[7] = acc1.w;
    if (lane == 0) lds_norm[wave] = norm_acc;
    __syncthreads();

    float ctsum = 0.f;
    #pragma unroll
    for (int w2 = 0; w2 < 8; ++w2) ctsum += lds_ct[w2][tid];
    partial_ct[((size_t)(b * kNBLK + blk)) * kH2 + tid] = ctsum;

    if (tid == 0) {
        float n = 0.f;
        #pragma unroll
        for (int w2 = 0; w2 < 8; ++w2) n += lds_norm[w2];
        partial_norm[b * kNBLK + blk] = n;
    }
}

// ---------------------------------------------------------------------------
// Kernel 3: per-batch finalize. Grid: kB blocks x 256 threads.
// norm[b] = sum of partials; ct_e = (sum partial_ct)/norm; alphat /= norm.
// ---------------------------------------------------------------------------
__global__ __launch_bounds__(256)
void attn_finalize(const float* __restrict__ partial_ct,
                   const float* __restrict__ partial_norm,
                   float* __restrict__ ct_out,     // (32,256)
                   float* __restrict__ alphat)     // (32,8192) in/out
{
    const int b = blockIdx.x;
    const int t = threadIdx.x;

    float norm = 0.f;
    #pragma unroll
    for (int k = 0; k < kNBLK; ++k) norm += partial_norm[b * kNBLK + k];
    const float inv = 1.0f / norm;

    float sum = 0.f;
    #pragma unroll
    for (int k = 0; k < kNBLK; ++k)
        sum += partial_ct[((size_t)(b * kNBLK + k)) * kH2 + t];
    ct_out[b * kH2 + t] = sum * inv;

    #pragma unroll
    for (int i = 0; i < kS / kH2; ++i) {
        const size_t idx = (size_t)b * kS + (size_t)i * kH2 + t;
        alphat[idx] *= inv;
    }
}

// ---------------------------------------------------------------------------
// Launch
// ---------------------------------------------------------------------------
extern "C" void kernel_launch(void* const* d_in, const int* in_sizes, int n_in,
                              void* d_out, int out_size, void* d_ws, size_t ws_size,
                              hipStream_t stream) {
    const float* h_d_t  = (const float*)d_in[0];   // (32,256)
    const float* h_enc  = (const float*)d_in[1];   // (32,8192,256)
    const float* mask   = (const float*)d_in[2];   // (32,8192)
    const float* sumexp = (const float*)d_in[3];   // (32,8192)
    const float* W      = (const float*)d_in[4];   // (256,256)

    float* out    = (float*)d_out;
    float* ct_out = out;                                   // 32*256
    float* alphat = out + kB * kH2;                        // 32*8192
    float* nsum   = out + kB * kH2 + kB * kS;              // 32*8192

    float* ws    = (float*)d_ws;
    float* q     = ws;                                     // 32*256
    float* pnorm = ws + kB * kH2;                          // 32*32
    float* pct   = ws + kB * kH2 + kB * kNBLK;             // 32*32*256

    qgemm_wmma<<<kB, 32, 0, stream>>>(h_d_t, W, q);

    dim3 g2(kNBLK, kB);
    attn_stream<<<g2, 256, 0, stream>>>(q, h_enc, mask, sumexp,
                                        alphat, nsum, pct, pnorm);

    attn_finalize<<<kB, 256, 0, stream>>>(pct, pnorm, ct_out, alphat);
}
